// InterRNN_22763326669359
// MI455X (gfx1250) — compile-verified, matrix operand order checked by reading, not run on hardware
//
#include <hip/hip_runtime.h>
#include <hip/hip_bf16.h>

#define H_  512
#define S_  100
#define B_  1024
#define NG_ 1536   // 3*H
#define BT_ 16     // batch rows per GRU block

typedef __attribute__((ext_vector_type(16))) __bf16 v16bf;
typedef __attribute__((ext_vector_type(8)))  __bf16 v8bf;
typedef __attribute__((ext_vector_type(8)))  float  v8f;

__device__ __forceinline__ v8f wmma_bf16(v16bf a, v16bf b, v8f c) {
  // emits v_wmma_f32_16x16x32_bf16
  return __builtin_amdgcn_wmma_f32_16x16x32_bf16(false, a, false, b, (short)0, c, false, false);
}

// A/B fragment loader, bf16 source. Per ISA 7.12.2 (16-bit A 16x32):
// lane<16: K {0..7} and {16..23}; lane>=16: K {8..15} and {24..31}.
// Caller passes p = row_base + kb + 8*(lane>>4); second segment at +16.
__device__ __forceinline__ v16bf frag_bf16(const __bf16* p) {
  v8bf lo = *(const v8bf*)(p);
  v8bf hi = *(const v8bf*)(p + 16);
  v16bf r;
#pragma unroll
  for (int i = 0; i < 8; ++i) { r[i] = lo[i]; r[8 + i] = hi[i]; }
  return r;
}

// Same fragment, fp32 source with inline bf16 conversion (for Wa gather).
__device__ __forceinline__ v16bf frag_f32(const float* p) {
  v16bf r;
#pragma unroll
  for (int i = 0; i < 8; ++i) r[i] = (__bf16)p[i];
#pragma unroll
  for (int i = 0; i < 8; ++i) r[8 + i] = (__bf16)p[16 + i];
  return r;
}

__device__ __forceinline__ float sigmoidf_(float x) { return 1.0f / (1.0f + __expf(-x)); }

// CDNA5 async global->LDS copy (ASYNCcnt-tracked, bypasses VGPRs).
__device__ __forceinline__ void async_ld_b128(unsigned lds_off, const void* g) {
  asm volatile("global_load_async_to_lds_b128 %0, %1, off"
               :: "v"(lds_off), "v"(g) : "memory");
}
__device__ __forceinline__ void wait_async0() {
#if __has_builtin(__builtin_amdgcn_s_wait_asynccnt)
  __builtin_amdgcn_s_wait_asynccnt(0);
#else
  asm volatile("s_wait_asynccnt 0x0" ::: "memory");
#endif
}

// ---------------------------------------------------------------------------
// K0a: convert W_ih / W_hh fp32 -> bf16 into workspace (contiguous: Wih | Whh)
// ---------------------------------------------------------------------------
__global__ void cvt_weights(const float* __restrict__ wih,
                            const float* __restrict__ whh,
                            __bf16* __restrict__ dst) {
  int i = blockIdx.x * 256 + threadIdx.x;
  const int N = NG_ * H_;
  if (i < N)           dst[i] = (__bf16)wih[i];
  else if (i < 2 * N)  dst[i] = (__bf16)whh[i - N];
}

// K0b: convert session representations fp32 -> bf16 (GRU A-matrix source)
__global__ void cvt_reps(const float* __restrict__ src, __bf16* __restrict__ dst) {
  int i = blockIdx.x * 256 + threadIdx.x;
  if (i < B_ * S_ * H_) dst[i] = (__bf16)src[i];
}

// ---------------------------------------------------------------------------
// K1: GRU recurrence. 64 blocks x 128 threads. Block owns 16 batch rows.
// Wave w computes h-column tiles j = w*8 .. w*8+7 (16 cols each -> H=512).
// Six accumulators per tile (x/h proj x r/z/n gates) -> gate math is
// per-lane elementwise; h kept in registers + bf16 LDS stage (double buffer).
// x_t tile staged via global_load_async_to_lds_b128 from the bf16 copy.
// ---------------------------------------------------------------------------
__global__ void __launch_bounds__(128) gru_kernel(
    const __bf16* __restrict__ Xbf,   // [B,S,H] bf16
    const float* __restrict__ h0,     // [B,H]   (hidden[0])
    const int*   __restrict__ cnt,    // [B]
    const float* __restrict__ b_ih,   // [3H]
    const float* __restrict__ b_hh,   // [3H]
    const __bf16* __restrict__ Wih,   // [3H,512] bf16
    const __bf16* __restrict__ Whh,   // [3H,512] bf16
    __bf16* __restrict__ outbf)       // [B,S,H] masked GRU output, bf16
{
  __shared__ __bf16 Xs[BT_][H_];        // 16 KB (async-load target)
  __shared__ __bf16 Hs[2][BT_][H_];     // 32 KB (double buffered)
  __shared__ float  bias[2 * NG_];      // 12 KB: b_ih | b_hh

  const int tid   = threadIdx.x;
  const int wave  = tid >> 5;
  const int lane  = tid & 31;
  const int hf    = lane >> 4;          // half-wave select
  const int lr    = lane & 15;
  const int koff  = 8 * hf;
  const int bbase = blockIdx.x * BT_;

  for (int i = tid; i < NG_; i += 128) { bias[i] = b_ih[i]; bias[NG_ + i] = b_hh[i]; }

  // h kept in registers: lane holds rows m=i+8*hf, cols hcol=16*j+lr, j=wave*8+jj
  float hold[8][8];
#pragma unroll
  for (int jj = 0; jj < 8; ++jj) {
    int hcol = 16 * (wave * 8 + jj) + lr;
#pragma unroll
    for (int i = 0; i < 8; ++i)
      hold[jj][i] = h0[(size_t)(bbase + i + 8 * hf) * H_ + hcol];
  }
  for (int i = tid; i < BT_ * H_; i += 128) {
    int m = i >> 9, h = i & (H_ - 1);
    Hs[0][m][h] = (__bf16)h0[(size_t)(bbase + m) * H_ + h];
  }
  int creg[8];
#pragma unroll
  for (int i = 0; i < 8; ++i) creg[i] = cnt[bbase + i + 8 * hf];

  __syncthreads();

  int cur = 0;
  for (int t = 0; t < S_; ++t) {
    // async-stage x_t (already bf16) for this block's 16 rows: 1024 b128 xfers
    {
      const __bf16* src = Xbf + ((size_t)bbase * S_ + t) * H_;
#pragma unroll
      for (int k = 0; k < 8; ++k) {
        int c   = tid + k * 128;        // chunk id 0..1023
        int m   = c >> 6;               // 64 chunks of 8 bf16 per row
        int col = (c & 63) * 8;
        unsigned ldsoff = (unsigned)(uintptr_t)&Xs[m][col];  // addr[31:0] = LDS addr
        async_ld_b128(ldsoff, src + (size_t)m * S_ * H_ + col);
      }
      wait_async0();
    }
    __syncthreads();
    const int nxt = cur ^ 1;

    const __bf16* xrow = &Xs[lr][0];
    const __bf16* hrow = &Hs[cur][lr][0];

#pragma unroll 1
    for (int jj = 0; jj < 8; ++jj) {
      const int hcol = 16 * (wave * 8 + jj) + lr;
      const __bf16* wir = Wih + (size_t)(hcol) * H_;
      const __bf16* wiz = Wih + (size_t)(H_ + hcol) * H_;
      const __bf16* win = Wih + (size_t)(2 * H_ + hcol) * H_;
      const __bf16* whr = Whh + (size_t)(hcol) * H_;
      const __bf16* whz = Whh + (size_t)(H_ + hcol) * H_;
      const __bf16* whn = Whh + (size_t)(2 * H_ + hcol) * H_;

      v8f axr = {0,0,0,0,0,0,0,0}, axz = axr, axn = axr;
      v8f ahr = axr, ahz = axr, ahn = axr;

#pragma unroll 4
      for (int kb = 0; kb < H_; kb += 32) {
        v16bf ax = frag_bf16(xrow + kb + koff);
        v16bf ah = frag_bf16(hrow + kb + koff);
        axr = wmma_bf16(ax, frag_bf16(wir + kb + koff), axr);
        axz = wmma_bf16(ax, frag_bf16(wiz + kb + koff), axz);
        axn = wmma_bf16(ax, frag_bf16(win + kb + koff), axn);
        ahr = wmma_bf16(ah, frag_bf16(whr + kb + koff), ahr);
        ahz = wmma_bf16(ah, frag_bf16(whz + kb + koff), ahz);
        ahn = wmma_bf16(ah, frag_bf16(whn + kb + koff), ahn);
      }

      const float bir = bias[hcol],        biz = bias[H_ + hcol],        bin = bias[2 * H_ + hcol];
      const float bhr = bias[NG_ + hcol],  bhz = bias[NG_ + H_ + hcol],  bhn = bias[NG_ + 2 * H_ + hcol];
#pragma unroll
      for (int i = 0; i < 8; ++i) {
        const int m = i + 8 * hf;
        float r  = sigmoidf_(axr[i] + bir + ahr[i] + bhr);
        float z  = sigmoidf_(axz[i] + biz + ahz[i] + bhz);
        float n  = tanhf(axn[i] + bin + r * (ahn[i] + bhn));
        float hn = (1.0f - z) * n + z * hold[jj][i];
        hold[jj][i] = hn;
        Hs[nxt][m][hcol] = (__bf16)hn;
        float ov = hn + ((t < creg[i]) ? 0.0f : -1.0e6f);   // mask for attention
        outbf[((size_t)(bbase + m) * S_ + t) * H_ + hcol] = (__bf16)ov;
      }
    }
    __syncthreads();
    cur = nxt;
  }
}

// ---------------------------------------------------------------------------
// K2: per-user routed attention. 1024 blocks (one per batch row), 128 thr.
// e = tanh(out[b] @ Wa[u]^T + ba[u]); k-outer loop with 7 accumulators so
// every Wa fragment is converted fp32->bf16 exactly once and reused across
// all 7 S-tiles. Scores folded via ds_add_f32; softmax; user_repr = attn@reps.
// bs[u] omitted: softmax shift-invariant.
// ---------------------------------------------------------------------------
__global__ void __launch_bounds__(128) attn_kernel(
    const __bf16* __restrict__ outbf,   // [B,S,H] bf16 masked GRU output
    const float*  __restrict__ Wa,      // [NU,H,H]
    const float*  __restrict__ ba,      // [NU,H]
    const float*  __restrict__ Ws,      // [NU,H]
    const float*  __restrict__ reps,    // [B,S,H] fp32
    const int*    __restrict__ ulist,   // [B]
    float* __restrict__ out)            // [1,B,H]
{
  __shared__ float sc[112];             // S padded to 7*16
  __shared__ float ba_s[H_];
  __shared__ float ws_s[H_];

  const int b    = blockIdx.x;
  const int tid  = threadIdx.x;
  const int wave = tid >> 5;
  const int lane = tid & 31;
  const int hf   = lane >> 4;
  const int lr   = lane & 15;
  const int koff = 8 * hf;
  const int u    = ulist[b];

  for (int i = tid; i < H_; i += 128) { ba_s[i] = ba[(size_t)u * H_ + i]; ws_s[i] = Ws[(size_t)u * H_ + i]; }
  for (int i = tid; i < 112; i += 128) sc[i] = 0.0f;
  __syncthreads();

  const float*  wa_u = Wa + (size_t)u * H_ * H_;
  const __bf16* arow = outbf + (size_t)b * S_ * H_;

#pragma unroll 1
  for (int q = 0; q < 8; ++q) {                       // wave's 8 output tiles
    const int n0   = (wave * 8 + q) * 16;
    const int o    = n0 + lr;                          // this lane's output col
    const float* wrow = wa_u + (size_t)o * H_;
    const float  ba_o = ba_s[o];
    const float  ws_o = ws_s[o];

    const __bf16* ar[7];
#pragma unroll
    for (int mt = 0; mt < 7; ++mt) {
      int srow = mt * 16 + lr; if (srow > S_ - 1) srow = S_ - 1;  // clamp pad rows
      ar[mt] = arow + (size_t)srow * H_;
    }

    v8f acc[7];
#pragma unroll
    for (int mt = 0; mt < 7; ++mt) acc[mt] = (v8f){0,0,0,0,0,0,0,0};

#pragma unroll 2
    for (int kb = 0; kb < H_; kb += 32) {
      v16bf w = frag_f32(wrow + kb + koff);            // converted ONCE per kb
#pragma unroll
      for (int mt = 0; mt < 7; ++mt)
        acc[mt] = wmma_bf16(frag_bf16(ar[mt] + kb + koff), w, acc[mt]);
    }

#pragma unroll
    for (int mt = 0; mt < 7; ++mt)
#pragma unroll
      for (int i = 0; i < 8; ++i) {
        float e = tanhf(acc[mt][i] + ba_o);
        atomicAdd(&sc[mt * 16 + i + 8 * hf], e * ws_o);  // ds_add_f32
      }
  }
  __syncthreads();

  if (tid == 0) {   // softmax over S=100 (tiny)
    float mx = -1e30f;
    for (int s = 0; s < S_; ++s) mx = fmaxf(mx, sc[s]);
    float sum = 0.0f;
    for (int s = 0; s < S_; ++s) { float e = __expf(sc[s] - mx); sc[s] = e; sum += e; }
    float inv = 1.0f / sum;
    for (int s = 0; s < S_; ++s) sc[s] *= inv;
  }
  __syncthreads();

  const float* rb = reps + (size_t)b * S_ * H_;
  for (int h = tid; h < H_; h += 128) {
    float acc = 0.0f;
    for (int s = 0; s < S_; ++s) acc += sc[s] * rb[(size_t)s * H_ + h];
    out[(size_t)b * H_ + h] = acc;
  }
}

// ---------------------------------------------------------------------------
extern "C" void kernel_launch(void* const* d_in, const int* in_sizes, int n_in,
                              void* d_out, int out_size, void* d_ws, size_t ws_size,
                              hipStream_t stream) {
  const float* reps  = (const float*)d_in[0];   // [B,S,H]
  const float* hid   = (const float*)d_in[1];   // [1,B,H]
  const int*   cnt   = (const int*)  d_in[2];   // [B]
  const int*   ulist = (const int*)  d_in[3];   // [B]
  const float* W_ih  = (const float*)d_in[4];   // [3H,H]
  const float* W_hh  = (const float*)d_in[5];   // [3H,H]
  const float* b_ih  = (const float*)d_in[6];   // [3H]
  const float* b_hh  = (const float*)d_in[7];   // [3H]
  const float* Wa    = (const float*)d_in[8];   // [NU,H,H]
  const float* ba    = (const float*)d_in[9];   // [NU,H]
  const float* Ws    = (const float*)d_in[10];  // [NU,H]
  const float* bs    = (const float*)d_in[11];  // [NU] (softmax-shift-invariant)
  (void)bs; (void)in_sizes; (void)n_in; (void)out_size; (void)ws_size;
  float* out = (float*)d_out;

  // workspace layout: Wih_bf16 | Whh_bf16 | Xbf16[B,S,H] | outbf16[B,S,H] (~213 MB)
  __bf16* wih_bf = (__bf16*)d_ws;
  __bf16* whh_bf = wih_bf + (size_t)NG_ * H_;
  __bf16* xbf    = whh_bf + (size_t)NG_ * H_;
  __bf16* outbf  = xbf    + (size_t)B_ * S_ * H_;

  cvt_weights<<<(2 * NG_ * H_ + 255) / 256, 256, 0, stream>>>(W_ih, W_hh, wih_bf);
  cvt_reps<<<(B_ * S_ * H_ + 255) / 256, 256, 0, stream>>>(reps, xbf);
  gru_kernel<<<B_ / BT_, 128, 0, stream>>>(xbf, hid, cnt, b_ih, b_hh, wih_bf, whh_bf, outbf);
  attn_kernel<<<B_, 128, 0, stream>>>(outbf, Wa, ba, Ws, reps, ulist, out);
}
